// SoftVectorQuantizing_82214263980159
// MI455X (gfx1250) — matrix operand order repaired and checked by reading, not compiled, not run on hardware
//
#include <hip/hip_runtime.h>

// SoftVectorQuantizing fused kernel for gfx1250 (MI455X, wave32, WMMA).
//
// x:      [65536, 256] f32        weight: [1024, 256] f32
// out:    quantized [65536,256] f32  ++  q_distribution [65536,1024] f32
//
// logits[m,n] = (2*(h . w_n) - |w_n|^2) / 256   (x_sq cancels in softmax)
// q = softmax(logits);  quantized = q @ W
//
// GEMMs run in bf16 via v_wmma_f32_16x16x32_bf16, f32 accumulate.
// f32->bf16 uses native hardware conversion (v_cvt_pk_bf16_f32).

typedef __bf16          v16bf  __attribute__((ext_vector_type(16)));
typedef float           v8f    __attribute__((ext_vector_type(8)));
typedef unsigned short  u16x16 __attribute__((ext_vector_type(16)));
typedef unsigned short  u16x8  __attribute__((ext_vector_type(8)));

#define NROWS 65536
#define DDIM  256
#define KDIM  1024
#define QOFF  ((size_t)NROWS * DDIM)   // offset of q_distribution in d_out

// f32 -> bf16 bits via native conversion (lowers to v_cvt_pk_bf16_f32)
static __device__ inline unsigned short f2bf(float f) {
  const __bf16 b = (__bf16)f;
  return __builtin_bit_cast(unsigned short, b);
}

// ---------------------------------------------------------------------------
// Prep: weight f32 -> bf16 row-major (wbf), bf16 transposed (wT), and |w|^2.
// 1024 threads, one codebook row each.
// ---------------------------------------------------------------------------
__global__ __launch_bounds__(256)
void svq_prep(const float* __restrict__ w,
              unsigned short* __restrict__ wbf,   // [1024, 256] bf16
              unsigned short* __restrict__ wT,    // [256, 1024] bf16
              float* __restrict__ wsq) {          // [1024]
  const int k = blockIdx.x * blockDim.x + threadIdx.x;
  if (k >= KDIM) return;
  const float* row = w + (size_t)k * DDIM;
  float s = 0.f;
  #pragma unroll 8
  for (int d = 0; d < DDIM; ++d) {
    const float v = row[d];
    s = fmaf(v, v, s);
    const unsigned short b = f2bf(v);
    wbf[(size_t)k * DDIM + d] = b;
    wT [(size_t)d * KDIM + k] = b;
  }
  wsq[k] = s;
}

// ---------------------------------------------------------------------------
// Main fused kernel: 256 threads (8 wave32) per block, 16 rows per block.
// ---------------------------------------------------------------------------
__global__ __launch_bounds__(256)
void svq_main(const float* __restrict__ x,
              const unsigned short* __restrict__ wbf,
              const unsigned short* __restrict__ wT,
              const float* __restrict__ wsq,
              float* __restrict__ out_q,    // quantized [65536,256]
              float* __restrict__ out_p) {  // q_distribution [65536,1024]
  __shared__ float dist[16 * KDIM];         // 64 KB logits tile.
                                            //  - first 8 KB: staged x (bf16)
                                            //    before GEMM1
                                            //  - low half of each row: bf16 q
                                            //    after softmax
  const int tid   = threadIdx.x;
  const int wid   = tid >> 5;
  const int lane  = tid & 31;
  const int nl    = lane & 15;              // M (A) / N (B,C) within tile
  const int h     = lane >> 4;              // lane-half selector
  const int mbase = blockIdx.x * 16;

  // ---- Stage x tile [16,256] f32 -> bf16 into LDS (once per block) ----
  {
    unsigned short* xs = (unsigned short*)dist;          // [16*256] bf16
    const float4* xsrc = (const float4*)(x + (size_t)mbase * DDIM) + tid * 4;
    #pragma unroll
    for (int c = 0; c < 4; ++c) {
      const float4 f = xsrc[c];
      unsigned short* p = xs + tid * 16 + c * 4;
      p[0] = f2bf(f.x); p[1] = f2bf(f.y);
      p[2] = f2bf(f.z); p[3] = f2bf(f.w);
    }
  }
  __syncthreads();

  // ---- GEMM1: wave owns columns [wid*128, wid*128+128) ----
  // kk-outer / tile-inner: one A fragment live (8 VGPRs), all 8 B fragments
  // for the chunk loaded up front (one big load clause -> staggered waits).
  v8f acc[8];
  #pragma unroll
  for (int t = 0; t < 8; ++t) acc[t] = v8f{0.f,0.f,0.f,0.f,0.f,0.f,0.f,0.f};
  {
    const unsigned short* xrow = (const unsigned short*)dist + nl * DDIM;
    const unsigned short* bbase = wbf + (size_t)(wid * 128 + nl) * DDIM;
    #pragma unroll
    for (int kk = 0; kk < 8; ++kk) {
      // A fragment for chunk kk from staged LDS x
      const int b0 = kk * 32 + 16 * h;      // elements 0..7  -> K = b0..b0+7
      const int b1 = b0 + 8;                // elements 8..15 -> K = b1..b1+7
      const u16x8 lo = *(const u16x8*)(xrow + b0);
      const u16x8 hi = *(const u16x8*)(xrow + b1);
      u16x16 a;
      #pragma unroll
      for (int e = 0; e < 8; ++e) { a[e] = lo[e]; a[e + 8] = hi[e]; }
      const v16bf av = __builtin_bit_cast(v16bf, a);
      // All 8 B fragments for this chunk
      u16x16 b[8];
      #pragma unroll
      for (int t = 0; t < 8; ++t)
        b[t] = *(const u16x16*)(bbase + (size_t)t * 16 * DDIM + kk * 32 + 16 * h);
      #pragma unroll
      for (int t = 0; t < 8; ++t)
        acc[t] = __builtin_amdgcn_wmma_f32_16x16x32_bf16(
                   false, av, false, __builtin_bit_cast(v16bf, b[t]),
                   (short)0, acc[t], false, false);
    }
  }
  __syncthreads();   // staged x fully consumed before dist is overwritten

  // Epilogue: logits into LDS
  #pragma unroll
  for (int t = 0; t < 8; ++t) {
    const int nbase = wid * 128 + t * 16;
    const float ws = wsq[nbase + nl];
    #pragma unroll
    for (int r = 0; r < 8; ++r) {
      const int m = r + 8 * h;              // C layout: M = r + 8*(lane/16)
      dist[m * KDIM + nbase + nl] = (2.0f * acc[t][r] - ws) * (1.0f / 256.0f);
    }
  }
  __syncthreads();

  // ---- Softmax over K=1024: wave handles rows 2*wid, 2*wid+1 ----
  #pragma unroll
  for (int rr = 0; rr < 2; ++rr) {
    const int row = wid * 2 + rr;
    float* drow = dist + row * KDIM;
    float v[32];
    #pragma unroll
    for (int i = 0; i < 32; ++i) v[i] = drow[lane + 32 * i];
    float mx = v[0];
    #pragma unroll
    for (int i = 1; i < 32; ++i) mx = fmaxf(mx, v[i]);
    #pragma unroll
    for (int off = 16; off > 0; off >>= 1)
      mx = fmaxf(mx, __shfl_xor(mx, off, 32));
    float s = 0.f;
    #pragma unroll
    for (int i = 0; i < 32; ++i) { v[i] = __expf(v[i] - mx); s += v[i]; }
    #pragma unroll
    for (int off = 16; off > 0; off >>= 1)
      s += __shfl_xor(s, off, 32);
    const float inv = 1.0f / s;
    float* prow = out_p + (size_t)(mbase + row) * KDIM;
    unsigned short* qb = (unsigned short*)(dist + row * KDIM);  // own row only
    #pragma unroll
    for (int i = 0; i < 32; ++i) {
      const float q = v[i] * inv;
      prow[lane + 32 * i] = q;            // f32 distribution out
      qb[lane + 32 * i]   = f2bf(q);      // bf16 A-operand for GEMM2
    }
  }
  __syncthreads();

  // ---- GEMM2: quantized[16,256] = q[16,1024] @ W[1024,256] ----
  // Process K chunks in pairs: hoist 2 A (LDS) + 4 B (L2) loads ahead of the
  // 4 WMMAs so the scheduler can overlap.
  const int d0 = wid * 32;
  const int d1 = wid * 32 + 16;
  v8f acc0 = {0.f,0.f,0.f,0.f,0.f,0.f,0.f,0.f};
  v8f acc1 = {0.f,0.f,0.f,0.f,0.f,0.f,0.f,0.f};
  const unsigned short* qrow =
      (const unsigned short*)dist + (size_t)nl * (KDIM * 2); // row stride 2048
  const unsigned short* wt0 = wT + (size_t)(d0 + nl) * KDIM;
  const unsigned short* wt1 = wT + (size_t)(d1 + nl) * KDIM;
  #pragma unroll 4
  for (int kp = 0; kp < 16; ++kp) {
    v16bf av[2];
    u16x16 bb0[2], bb1[2];
    #pragma unroll
    for (int u = 0; u < 2; ++u) {
      const int kk = kp * 2 + u;
      const int b0 = kk * 32 + 16 * h;
      const int b1 = b0 + 8;
      const u16x8 lo = *(const u16x8*)(qrow + b0);
      const u16x8 hi = *(const u16x8*)(qrow + b1);
      u16x16 a;
      #pragma unroll
      for (int e = 0; e < 8; ++e) { a[e] = lo[e]; a[e + 8] = hi[e]; }
      av[u]  = __builtin_bit_cast(v16bf, a);
      bb0[u] = *(const u16x16*)(wt0 + kk * 32 + 16 * h);
      bb1[u] = *(const u16x16*)(wt1 + kk * 32 + 16 * h);
    }
    #pragma unroll
    for (int u = 0; u < 2; ++u) {
      acc0 = __builtin_amdgcn_wmma_f32_16x16x32_bf16(
               false, av[u], false, __builtin_bit_cast(v16bf, bb0[u]),
               (short)0, acc0, false, false);
      acc1 = __builtin_amdgcn_wmma_f32_16x16x32_bf16(
               false, av[u], false, __builtin_bit_cast(v16bf, bb1[u]),
               (short)0, acc1, false, false);
    }
  }
  #pragma unroll
  for (int r = 0; r < 8; ++r) {
    const int m = mbase + r + 8 * h;
    out_q[(size_t)m * DDIM + d0 + nl] = acc0[r];
    out_q[(size_t)m * DDIM + d1 + nl] = acc1[r];
  }
}

// ---------------------------------------------------------------------------
extern "C" void kernel_launch(void* const* d_in, const int* in_sizes, int n_in,
                              void* d_out, int out_size, void* d_ws, size_t ws_size,
                              hipStream_t stream) {
  const float* x = (const float*)d_in[0];      // [16,4096,256]
  const float* w = (const float*)d_in[1];      // [1024,256]

  unsigned short* wbf = (unsigned short*)d_ws;              // 512 KB
  unsigned short* wT  = wbf + (size_t)KDIM * DDIM;          // 512 KB
  float*          wsq = (float*)(wT + (size_t)DDIM * KDIM); // 4 KB

  float* out = (float*)d_out;

  svq_prep<<<KDIM / 256, 256, 0, stream>>>(w, wbf, wT, wsq);
  svq_main<<<NROWS / 16, 256, 0, stream>>>(x, wbf, wT, wsq, out, out + QOFF);
}
